// LSTMUnit_56444460204062
// MI455X (gfx1250) — matrix operand order repaired and checked by reading, not compiled, not run on hardware
//
#include <hip/hip_runtime.h>
#include <cstdint>
#include <cstddef>

// ---------------------------------------------------------------------------
// LSTM on gfx1250 (MI455X): bf16 WMMA recurrence, one kernel per time step.
//   gates = [x_t | h] @ W + b   (K = E+H = 1536 fused GEMM, M=64, N=4096)
// Block = one 16x16 M-N tile replicated over the 4 gate blocks.
// 4 waves split K (12 k-tiles each), LDS reduction, wave 0 runs the fused
// sigmoid/tanh + c/h update epilogue. 2-deep software-pipelined fragment
// loads; all fragment addresses are (uniform tile base) + (lane*32B) so the
// backend can use scalar-base (saddr) global loads and SALU address updates.
// ---------------------------------------------------------------------------

typedef __attribute__((ext_vector_type(16))) __bf16 v16bf;
typedef __attribute__((ext_vector_type(8)))  float  v8f;

constexpr int NB = 64, NT = 512, NE = 512, NH = 1024;
constexpr int KTOT   = NE + NH;   // 1536 (rows 0..511 = Wx, 512..1535 = Wh)
constexpr int KT_X   = NE / 32;   // 16 k-tiles from x
constexpr int KT_H   = NH / 32;   // 32 k-tiles from h
constexpr int KT_ALL = KTOT / 32; // 48
constexpr int KQ     = KT_ALL / 4;// 12 k-tiles per K-quarter wave
constexpr int NTILES_G = NH / 16; // 64 column tiles per gate
constexpr int MTILES   = NB / 16; // 4

// workspace layout (bytes, all chunks multiple of 256)
constexpr size_t WT_OFF   = 0;
constexpr size_t WT_BYTES = (size_t)4 * NH * KTOT * 2;    // 12,582,912 bf16 W, fragment order
constexpr size_t XF_OFF   = WT_OFF + WT_BYTES;
constexpr size_t XF_BYTES = (size_t)NB * NT * NE * 2;     // 33,554,432 bf16 x, fragment order
constexpr size_t HF_OFF   = XF_OFF + XF_BYTES;
constexpr size_t HF_BYTES = (size_t)2 * NB * NH * 2;      // 262,144 bf16 h, double buffered
constexpr size_t CS_OFF   = HF_OFF + HF_BYTES;            // + 262,144 fp32 c state

__device__ __forceinline__ float sigmoidf_(float x) {
    return 1.0f / (1.0f + __expf(-x));
}

// fragment-order index for an element (row m, k) of a 16(M)x32(K) bf16 A-tile:
// lane = (m&15) | (half<<4); half selects K-subranges {0-7,16-23} vs {8-15,24-31}
__device__ __forceinline__ size_t hfrag_index(int m, int hidx) {
    int mt   = m >> 4;
    int kt   = hidx >> 5;
    int k32  = hidx & 31;
    int half = (k32 >> 3) & 1;
    int e    = (k32 & 7) + ((k32 & 16) ? 8 : 0);
    int lane = (m & 15) | (half << 4);
    return ((size_t)(mt * KT_H + kt) << 9) + ((size_t)lane << 4) + e;
}

// ---- one-time converts -----------------------------------------------------

// W fp32 [KTOT][4H] -> bf16 fragment-order B tiles:
// tile id = ng*48 + kt (ng = 16-wide column tile over 4H), 512 elems per tile,
// lane = n%16 | (khalf<<4), per-lane 16 elems = K {0-7,16-23}/{8-15,24-31} (B layout).
__global__ void lstm_cvt_w(const float* __restrict__ W, __bf16* __restrict__ wf) {
    int i = blockIdx.x * blockDim.x + threadIdx.x;
    if (i >= 4 * NH * KTOT) return;
    int tile = i >> 9;
    int j    = i & 511;
    int lane = j >> 4, e = j & 15;
    int ng = tile / KT_ALL;
    int kt = tile % KT_ALL;
    int n  = ng * 16 + (lane & 15);
    int k  = kt * 32 + (lane >> 4) * 8 + (e < 8 ? e : e + 8);
    wf[i] = (__bf16)W[(size_t)k * (4 * NH) + n];
}

// x fp32 [B][T][E] -> bf16 fragment-order A tiles:
// tile id = ((mt*NT + t)*KT_X + kt), per-lane 16 elems = K {0-7,16-23}/{8-15,24-31}
__global__ void lstm_cvt_x(const float* __restrict__ x, __bf16* __restrict__ xf) {
    int i = blockIdx.x * blockDim.x + threadIdx.x;
    if (i >= NB * NT * NE) return;
    int tile = i >> 9;
    int j    = i & 511;
    int lane = j >> 4, e = j & 15;
    int mt  = tile / (NT * KT_X);
    int rem = tile % (NT * KT_X);
    int t   = rem / KT_X;
    int kt  = rem % KT_X;
    int m    = mt * 16 + (lane & 15);
    int half = lane >> 4;
    int k    = kt * 32 + half * 8 + (e < 8 ? e : e + 8);
    xf[i] = (__bf16)x[((size_t)m * NT + t) * NE + k];
}

__global__ void lstm_init(const float* __restrict__ h0, const float* __restrict__ c0,
                          __bf16* __restrict__ hf0, float* __restrict__ cs) {
    int i = blockIdx.x * blockDim.x + threadIdx.x;
    if (i >= NB * NH) return;
    int m = i / NH, hh = i % NH;
    hf0[hfrag_index(m, hh)] = (__bf16)h0[i];
    cs[i] = c0[i];
}

// ---- per-time-step fused split-K GEMM + gate kernel ------------------------

#define WMMA_BF16(A_, B_, C_) \
    __builtin_amdgcn_wmma_f32_16x16x32_bf16(false, (A_), false, (B_), (short)0, (C_), false, false)

__global__ __launch_bounds__(128) void lstm_step(
    const __bf16* __restrict__ xf, const __bf16* __restrict__ wf,
    const float* __restrict__ bias,
    const __bf16* __restrict__ hin, __bf16* __restrict__ hout,
    float* __restrict__ cs, float* __restrict__ out, int t)
{
    __shared__ v8f red[3 * 4 * 32];   // partials of waves kq=1..3, 4 gates, 32 lanes

    const int lane  = threadIdx.x & 31;
    // kq is wave-uniform: force it scalar so all kt / tile-base math stays in SGPRs
    const int kq    = __builtin_amdgcn_readfirstlane((int)threadIdx.x) >> 5;  // 0..3
    const int mtile = blockIdx.x & (MTILES - 1);   // 0..3
    const int ntile = blockIdx.x >> 2;             // 0..63: 16-col tile within each gate
    const int half  = lane >> 4;
    const int nn    = ntile * 16 + (lane & 15);    // this lane's column within a gate
    const int loff  = lane << 4;                   // per-lane element offset (32 B)

    // uniform (scalar) tile base pointers; every load is base + loff
    const __bf16* axu = xf  + (((size_t)(mtile * NT + t) * KT_X) << 9);
    const __bf16* ahu = hin + (((size_t)mtile * KT_H) << 9);
    const __bf16* bu0 = wf + (((size_t)(0 * NTILES_G + ntile) * KT_ALL) << 9);
    const __bf16* bu1 = wf + (((size_t)(1 * NTILES_G + ntile) * KT_ALL) << 9);
    const __bf16* bu2 = wf + (((size_t)(2 * NTILES_G + ntile) * KT_ALL) << 9);
    const __bf16* bu3 = wf + (((size_t)(3 * NTILES_G + ntile) * KT_ALL) << 9);

    const int kt0 = kq * KQ;

    auto atile = [&](int kt) -> const __bf16* {    // uniform A tile base: x part then h part
        return (kt < KT_X) ? (axu + ((size_t)kt << 9))
                           : (ahu + ((size_t)(kt - KT_X) << 9));
    };

    v8f ai = {}, af = {}, ag = {}, ao = {};

    // 2-deep software pipeline: loads for k-step i+2 issue before WMMAs of i.
    v16bf A[2], B0[2], B1[2], B2[2], B3[2];
#pragma unroll
    for (int p = 0; p < 2; ++p) {
        int kt = kt0 + p;
        A[p]  = *(const v16bf*)(atile(kt) + loff);
        B0[p] = *(const v16bf*)(bu0 + ((size_t)kt << 9) + loff);
        B1[p] = *(const v16bf*)(bu1 + ((size_t)kt << 9) + loff);
        B2[p] = *(const v16bf*)(bu2 + ((size_t)kt << 9) + loff);
        B3[p] = *(const v16bf*)(bu3 + ((size_t)kt << 9) + loff);
    }
#pragma unroll
    for (int i = 0; i < KQ; ++i) {
        v16bf an = {}, b0n = {}, b1n = {}, b2n = {}, b3n = {};
        if (i + 2 < KQ) {
            int kt = kt0 + i + 2;
            an  = *(const v16bf*)(atile(kt) + loff);
            b0n = *(const v16bf*)(bu0 + ((size_t)kt << 9) + loff);
            b1n = *(const v16bf*)(bu1 + ((size_t)kt << 9) + loff);
            b2n = *(const v16bf*)(bu2 + ((size_t)kt << 9) + loff);
            b3n = *(const v16bf*)(bu3 + ((size_t)kt << 9) + loff);
        }
        const int c = i & 1;
        ai = WMMA_BF16(A[c], B0[c], ai);
        af = WMMA_BF16(A[c], B1[c], af);
        ag = WMMA_BF16(A[c], B2[c], ag);
        ao = WMMA_BF16(A[c], B3[c], ao);
        A[c] = an; B0[c] = b0n; B1[c] = b1n; B2[c] = b2n; B3[c] = b3n;
    }

    // split-K reduction through LDS
    if (kq != 0) {
        const int base = (kq - 1) * 4 * 32 + lane;
        red[base +  0] = ai;
        red[base + 32] = af;
        red[base + 64] = ag;
        red[base + 96] = ao;
    }
    __syncthreads();
    if (kq != 0) return;

#pragma unroll
    for (int w = 0; w < 3; ++w) {
        const int base = w * 4 * 32 + lane;
        ai += red[base +  0];
        af += red[base + 32];
        ag += red[base + 64];
        ao += red[base + 96];
    }

    const float bI = bias[0 * NH + nn];
    const float bF = bias[1 * NH + nn] + 1.0f;   // reference: sigmoid(f + 1.0)
    const float bG = bias[2 * NH + nn];
    const float bO = bias[3 * NH + nn];

    // C/D layout: VGPR v -> (M = v + 8*half, N = lane&15) within the tile
#pragma unroll
    for (int v = 0; v < 8; ++v) {
        int m = mtile * 16 + v + half * 8;
        float ig = sigmoidf_(ai[v] + bI);
        float fg = sigmoidf_(af[v] + bF);
        float gg = tanhf(ag[v] + bG);
        float og = sigmoidf_(ao[v] + bO);
        size_t ci = (size_t)m * NH + nn;          // (m,nn) owned by exactly this lane
        float c = fg * cs[ci] + ig * gg;
        cs[ci] = c;
        float h = og * tanhf(c);
        hout[hfrag_index(m, nn)] = (__bf16)h;     // fragment order for next step's A
        out[((size_t)m * NT + t) * NH + nn] = h;
    }
}

__global__ void lstm_final(const float* __restrict__ out, const float* __restrict__ cs,
                           float* __restrict__ hT, float* __restrict__ cT) {
    int i = blockIdx.x * blockDim.x + threadIdx.x;
    if (i >= NB * NH) return;
    int m = i / NH, hh = i % NH;
    hT[i] = out[((size_t)m * NT + (NT - 1)) * NH + hh];
    cT[i] = cs[i];
}

// ---------------------------------------------------------------------------

extern "C" void kernel_launch(void* const* d_in, const int* /*in_sizes*/, int /*n_in*/,
                              void* d_out, int /*out_size*/, void* d_ws, size_t /*ws_size*/,
                              hipStream_t stream) {
    const float* x  = (const float*)d_in[0];
    const float* h0 = (const float*)d_in[1];
    const float* c0 = (const float*)d_in[2];
    const float* W  = (const float*)d_in[3];
    const float* b  = (const float*)d_in[4];
    float* out = (float*)d_out;

    char* ws = (char*)d_ws;
    __bf16* wf = (__bf16*)(ws + WT_OFF);
    __bf16* xf = (__bf16*)(ws + XF_OFF);
    __bf16* hf = (__bf16*)(ws + HF_OFF);
    float*  cs = (float*)(ws + CS_OFF);

    lstm_cvt_w<<<(4 * NH * KTOT + 255) / 256, 256, 0, stream>>>(W, wf);
    lstm_cvt_x<<<(NB * NT * NE + 255) / 256, 256, 0, stream>>>(x, xf);
    lstm_init<<<(NB * NH + 255) / 256, 256, 0, stream>>>(h0, c0, hf, cs);

    const size_t HBUF = (size_t)NB * NH;  // bf16 elements per h buffer
    for (int t = 0; t < NT; ++t) {
        const __bf16* hin  = hf + (size_t)(t & 1) * HBUF;
        __bf16*       hout = hf + (size_t)((t + 1) & 1) * HBUF;
        lstm_step<<<MTILES * NTILES_G, 128, 0, stream>>>(
            xf, wf, b, hin, hout, cs, out, t);
    }

    lstm_final<<<(NB * NH + 255) / 256, 256, 0, stream>>>(
        out, cs, out + (size_t)NB * NT * NH, out + (size_t)NB * NT * NH + (size_t)NB * NH);
}
